// encoder_srnn_21388937134091
// MI455X (gfx1250) — matrix operand order repaired and checked by reading, not compiled, not run on hardware
//
#include <hip/hip_runtime.h>
#include <hip/hip_fp16.h>
#include <stdint.h>

// ---------------------------------------------------------------------------
// CDNA5 (gfx1250) implementation of the SRNN encoder reference.
//  - batched input projections  : v_wmma_f32_16x16x32_f16 (f32 accumulate),
//    2x2 tile blocking per wave (32x32 output), 8 b128 loads -> 4 WMMAs / k-step
//  - sequential LSTM recurrence : 1 workgroup / direction, h in LDS,
//    f16 Whh streamed from L2 via b128 loads
// ---------------------------------------------------------------------------

typedef __attribute__((ext_vector_type(16))) _Float16 v16h;
typedef __attribute__((ext_vector_type(8)))  _Float16 v8h;
typedef __attribute__((ext_vector_type(8)))  float    v8f;

#define HDIM  512
#define G4    2048          // 4*HDIM gate width
#define TLEN  4096
#define SLEN  128
#define NSEG  1920          // copy_rep segments

// ---------------- f32 -> f16 conversion (RNE via v_cvt_f16_f32) -------------
__global__ void k_cvt_f16(const float* __restrict__ src,
                          _Float16* __restrict__ dst, int n) {
  int i = blockIdx.x * blockDim.x + threadIdx.x;
  int stride = gridDim.x * blockDim.x;
  for (; i < n; i += stride) dst[i] = (_Float16)src[i];
}

__device__ __forceinline__ v16h cat8(v8h lo, v8h hi) {
  return __builtin_shufflevector(lo, hi, 0, 1, 2, 3, 4, 5, 6, 7,
                                         8, 9, 10, 11, 12, 13, 14, 15);
}

// ---------------- WMMA GEMM:  C[M,N] = A[M,K] @ W[N,K]^T + bias1 + bias2 ----
// One wave owns a 32x32 output block: 2 A tiles x 2 B tiles -> 4 WMMA tiles.
// All 8 b128 loads of a k-step are issued before the 4 WMMAs so the scheduler
// can clause them and overlap WMMA issue with outstanding loads.
// A operand layout (16-bit 16x32): lane<16 holds K 0..7 & 16..23 of row M=lane,
//   lane>=16 holds K 8..15 & 24..31 of the same row.
// B operand layout (16-bit 32x16): lane<16 = column N=lane, K 0..7 & 16..23;
//   lane>=16 = same column, K 8..15 & 24..31.   (CDNA5 ISA §7.12.2)
__global__ void __launch_bounds__(256)
k_gemm_f16_wmma(const _Float16* __restrict__ A, const _Float16* __restrict__ W,
                const float* __restrict__ bias1, const float* __restrict__ bias2,
                float* __restrict__ C, int M, int N, int K) {
  const int wid  = (blockIdx.x * 256 + threadIdx.x) >> 5;
  const int lane = threadIdx.x & 31;
  const int tiles_n = N >> 5;              // 32-wide column strips
  const int tiles_m = M >> 5;              // 32-tall row strips
  if (wid >= tiles_m * tiles_n) return;    // wave-uniform: EXEC stays all-ones
  const int tm  = (wid / tiles_n) << 5;
  const int tn  = (wid % tiles_n) << 5;
  const int r   = lane & 15;
  const int hiL = (lane & 16) ? 1 : 0;
  const int kloA = hiL ? 8  : 0;
  const int kloB = hiL ? 8 : 0;
  const _Float16* arow0 = A + (size_t)(tm + r) * K;
  const _Float16* arow1 = A + (size_t)(tm + 16 + r) * K;
  const _Float16* wrow0 = W + (size_t)(tn + r) * K;
  const _Float16* wrow1 = W + (size_t)(tn + 16 + r) * K;

  v8f acc00 = {}, acc01 = {}, acc10 = {}, acc11 = {};
  for (int k0 = 0; k0 < K; k0 += 32) {
    // issue all loads for this k-step first (distinct registers)
    v8h a0l = *reinterpret_cast<const v8h*>(arow0 + k0 + kloA);
    v8h a0h = *reinterpret_cast<const v8h*>(arow0 + k0 + 16 + kloA);
    v8h a1l = *reinterpret_cast<const v8h*>(arow1 + k0 + kloA);
    v8h a1h = *reinterpret_cast<const v8h*>(arow1 + k0 + 16 + kloA);
    v8h b0l = *reinterpret_cast<const v8h*>(wrow0 + k0 + kloB);
    v8h b0h = *reinterpret_cast<const v8h*>(wrow0 + k0 + 16 + kloB);
    v8h b1l = *reinterpret_cast<const v8h*>(wrow1 + k0 + kloB);
    v8h b1h = *reinterpret_cast<const v8h*>(wrow1 + k0 + 16 + kloB);
    v16h a0 = cat8(a0l, a0h);
    v16h a1 = cat8(a1l, a1h);
    v16h b0 = cat8(b0l, b0h);
    v16h b1 = cat8(b1l, b1h);
    acc00 = __builtin_amdgcn_wmma_f32_16x16x32_f16(false, a0, false, b0,
                                                   (short)0, acc00, false, false);
    acc01 = __builtin_amdgcn_wmma_f32_16x16x32_f16(false, a0, false, b1,
                                                   (short)0, acc01, false, false);
    acc10 = __builtin_amdgcn_wmma_f32_16x16x32_f16(false, a1, false, b0,
                                                   (short)0, acc10, false, false);
    acc11 = __builtin_amdgcn_wmma_f32_16x16x32_f16(false, a1, false, b1,
                                                   (short)0, acc11, false, false);
  }
  // D layout: VGPR v -> M = v (+8 for lanes 16..31), N = lane&15 within tile
  const int mb0 = tm + (hiL ? 8 : 0);
  const int mb1 = mb0 + 16;
  const int n0  = tn + r;
  const int n1  = n0 + 16;
  const float bs0 = (bias1 ? bias1[n0] : 0.0f) + (bias2 ? bias2[n0] : 0.0f);
  const float bs1 = (bias1 ? bias1[n1] : 0.0f) + (bias2 ? bias2[n1] : 0.0f);
#pragma unroll
  for (int v = 0; v < 8; ++v) {
    C[(size_t)(mb0 + v) * N + n0] = acc00[v] + bs0;
    C[(size_t)(mb0 + v) * N + n1] = acc01[v] + bs1;
    C[(size_t)(mb1 + v) * N + n0] = acc10[v] + bs0;
    C[(size_t)(mb1 + v) * N + n1] = acc11[v] + bs1;
  }
}

// ---------------- sequential LSTM recurrence --------------------------------
// grid = 2 blocks (block 0: forward, block 1: backward), 1024 threads.
// XP holds x@Wih^T + bih + bhh precomputed.  Thread tid computes gate
// pre-activations j=tid and j=tid+1024; threads < HDIM own cell k=tid.
// Gate order (torch): i | f | g | o.
__global__ void __launch_bounds__(1024)
k_lstm_recur(const float* __restrict__ XPf, const float* __restrict__ XPb,
             const _Float16* __restrict__ Whhf, const _Float16* __restrict__ Whhb,
             float* __restrict__ out, _Float16* __restrict__ out_h,
             float* __restrict__ hT, float* __restrict__ cT, int T) {
  __shared__ __align__(16) float h_sh[HDIM];
  __shared__ float g_sh[G4];
  const int dir = blockIdx.x;
  const float*    XP = dir ? XPb  : XPf;
  const _Float16* Wu = dir ? Whhb : Whhf;
  const int colOff   = dir ? HDIM : 0;
  const int tid = threadIdx.x;
  if (tid < HDIM) h_sh[tid] = 0.0f;
  float c = 0.0f;
  __syncthreads();
  const v8h* W0 = reinterpret_cast<const v8h*>(Wu + (size_t)tid * HDIM);
  const v8h* W1 = reinterpret_cast<const v8h*>(Wu + (size_t)(tid + 1024) * HDIM);

  for (int s = 0; s < T; ++s) {
    const int t = dir ? (T - 1 - s) : s;
    const float* xp = XP + (size_t)t * G4;
    float a0 = xp[tid], a1 = xp[tid + 1024];
    const float4* hp = reinterpret_cast<const float4*>(h_sh);
#pragma unroll 4
    for (int i = 0; i < HDIM / 8; ++i) {
      v8h w0 = W0[i], w1 = W1[i];
      float4 h0 = hp[2 * i], h1 = hp[2 * i + 1];
      a0 += (float)w0[0]*h0.x + (float)w0[1]*h0.y + (float)w0[2]*h0.z + (float)w0[3]*h0.w
          + (float)w0[4]*h1.x + (float)w0[5]*h1.y + (float)w0[6]*h1.z + (float)w0[7]*h1.w;
      a1 += (float)w1[0]*h0.x + (float)w1[1]*h0.y + (float)w1[2]*h0.z + (float)w1[3]*h0.w
          + (float)w1[4]*h1.x + (float)w1[5]*h1.y + (float)w1[6]*h1.z + (float)w1[7]*h1.w;
    }
    g_sh[tid] = a0;
    g_sh[tid + 1024] = a1;
    __syncthreads();
    if (tid < HDIM) {
      const float gi = 1.0f / (1.0f + __expf(-g_sh[tid]));
      const float gf = 1.0f / (1.0f + __expf(-g_sh[tid + 512]));
      const float gg = tanhf(g_sh[tid + 1024]);
      const float go = 1.0f / (1.0f + __expf(-g_sh[tid + 1536]));
      c = gf * c + gi * gg;
      const float h = go * tanhf(c);
      h_sh[tid] = h;
      out[(size_t)t * (2 * HDIM) + colOff + tid] = h;
      if (out_h) out_h[(size_t)t * (2 * HDIM) + colOff + tid] = (_Float16)h;
    }
    __syncthreads();
  }
  if (tid < HDIM) {
    hT[dir * HDIM + tid] = h_sh[tid];
    cT[dir * HDIM + tid] = c;
  }
}

// ---------------- sentence-input gather (fwd@end, bwd@start) -> f16 ---------
__global__ void k_sent_gather(const float* __restrict__ out,
                              const int* __restrict__ seps,
                              _Float16* __restrict__ dst) {
  const int s = blockIdx.x;       // 0..SLEN-1
  const int d = threadIdx.x;      // 0..1023
  const float v = (d < HDIM) ? out[(size_t)seps[s + 1] * (2 * HDIM) + d]
                             : out[(size_t)seps[s]     * (2 * HDIM) + d];
  dst[(size_t)s * (2 * HDIM) + d] = (_Float16)v;
}

// ---------------- copy_rep: segment mean (comb_seg[e] == e/2 by construction)
__global__ void k_copy_rep(const float* __restrict__ out,
                           const int* __restrict__ comb_idx,
                           const float* __restrict__ counts,
                           float* __restrict__ cr) {
  int i = blockIdx.x * blockDim.x + threadIdx.x;
  const int total = NSEG * (2 * HDIM);
  const int stride = gridDim.x * blockDim.x;
  for (; i < total; i += stride) {
    const int g = i >> 10;
    const int d = i & 1023;
    const int p0 = comb_idx[2 * g], p1 = comb_idx[2 * g + 1];
    cr[i] = (out[(size_t)p0 * (2 * HDIM) + d] +
             out[(size_t)p1 * (2 * HDIM) + d]) / counts[g];
  }
}

// ---------------------------------------------------------------------------
extern "C" void kernel_launch(void* const* d_in, const int* in_sizes, int n_in,
                              void* d_out, int out_size, void* d_ws, size_t ws_size,
                              hipStream_t stream) {
  (void)in_sizes; (void)n_in; (void)out_size; (void)ws_size;
  // input ordering (pytree flatten of setup_inputs dict):
  // 0 input_t, 1 seps, 2 comb_idx, 3 comb_seg, 4 comb_counts,
  // 5..20 lstm_params (l0f,l0b,l1f,l1b) x (Wih,Whh,bih,bhh),
  // 21 w2s_W, 22 w2s_b, 23..38 sent_lstm_params (same order)
  const float* x    = (const float*)d_in[0];
  const int*   seps = (const int*)d_in[1];
  const int*   cidx = (const int*)d_in[2];
  const float* ccnt = (const float*)d_in[4];
  const float *Wih[4], *Whh[4], *bih[4], *bhh[4];
  const float *sWih[4], *sWhh[4], *sbih[4], *sbhh[4];
  for (int i = 0; i < 4; ++i) {
    Wih[i]  = (const float*)d_in[5  + 4 * i];
    Whh[i]  = (const float*)d_in[6  + 4 * i];
    bih[i]  = (const float*)d_in[7  + 4 * i];
    bhh[i]  = (const float*)d_in[8  + 4 * i];
    sWih[i] = (const float*)d_in[23 + 4 * i];
    sWhh[i] = (const float*)d_in[24 + 4 * i];
    sbih[i] = (const float*)d_in[25 + 4 * i];
    sbhh[i] = (const float*)d_in[26 + 4 * i];
  }
  const float* w2sW = (const float*)d_in[21];
  const float* w2sb = (const float*)d_in[22];

  // workspace bump allocator (256B aligned slots)
  char* base = (char*)d_ws;
  size_t off = 0;
  auto alloc = [&](size_t elems, size_t esz) -> void* {
    void* p = base + off;
    off += (elems * esz + 255) & ~(size_t)255;
    return p;
  };
  _Float16* x_h = (_Float16*)alloc((size_t)TLEN * 512, 2);
  const size_t wih_sz[4] = {2048ull * 512, 2048ull * 512,
                            2048ull * 1024, 2048ull * 1024};
  _Float16 *Wih_h[4], *Whh_h[4], *sWih_h[4], *sWhh_h[4];
  for (int i = 0; i < 4; ++i) {
    Wih_h[i]  = (_Float16*)alloc(wih_sz[i], 2);
    Whh_h[i]  = (_Float16*)alloc(2048ull * 512, 2);
    sWih_h[i] = (_Float16*)alloc(wih_sz[i], 2);
    sWhh_h[i] = (_Float16*)alloc(2048ull * 512, 2);
  }
  _Float16* w2sW_h  = (_Float16*)alloc(512ull * 1024, 2);
  float*    XPf     = (float*)alloc((size_t)TLEN * G4, 4);   // reused per layer
  float*    XPb     = (float*)alloc((size_t)TLEN * G4, 4);
  float*    out0    = (float*)alloc((size_t)TLEN * 1024, 4);
  _Float16* out0_h  = (_Float16*)alloc((size_t)TLEN * 1024, 2);
  _Float16* sg_h    = (_Float16*)alloc((size_t)SLEN * 1024, 2);
  float*    sin_f   = (float*)alloc((size_t)SLEN * 512, 4);
  _Float16* sin_h   = (_Float16*)alloc((size_t)SLEN * 512, 2);
  float*    sout0   = (float*)alloc((size_t)SLEN * 1024, 4);
  _Float16* sout0_h = (_Float16*)alloc((size_t)SLEN * 1024, 2);
  float*    dump    = (float*)alloc(2048, 4);  // sent-LSTM hT/cT (discarded)

  float* O         = (float*)d_out;
  float* out_final = O;              // [4096,1024]
  float* sent_out  = O + 4194304;    // [128,1024]
  float* crep      = O + 4325376;    // [1920,1024]
  float* hid_h     = O + 6291456;    // [4,512]  (l0f,l0b,l1f,l1b)
  float* hid_c     = O + 6293504;    // [4,512]

  auto cvt = [&](const float* s, _Float16* d, int n) {
    int blocks = (n + 255) / 256;
    if (blocks > 4096) blocks = 4096;
    k_cvt_f16<<<blocks, 256, 0, stream>>>(s, d, n);
  };
  auto gemm = [&](const _Float16* A, const _Float16* W, const float* b1,
                  const float* b2, float* C, int M, int N, int K) {
    int waves = (M / 32) * (N / 32);
    k_gemm_f16_wmma<<<(waves + 7) / 8, 256, 0, stream>>>(A, W, b1, b2, C, M, N, K);
  };

  // ---- precision conversions (recomputed each call; deterministic) --------
  cvt(x, x_h, TLEN * 512);
  for (int i = 0; i < 4; ++i) {
    cvt(Wih[i],  Wih_h[i],  (int)wih_sz[i]);
    cvt(Whh[i],  Whh_h[i],  2048 * 512);
    cvt(sWih[i], sWih_h[i], (int)wih_sz[i]);
    cvt(sWhh[i], sWhh_h[i], 2048 * 512);
  }
  cvt(w2sW, w2sW_h, 512 * 1024);

  // ---- main biLSTM layer 0 ------------------------------------------------
  gemm(x_h, Wih_h[0], bih[0], bhh[0], XPf, TLEN, G4, 512);
  gemm(x_h, Wih_h[1], bih[1], bhh[1], XPb, TLEN, G4, 512);
  k_lstm_recur<<<2, 1024, 0, stream>>>(XPf, XPb, Whh_h[0], Whh_h[1],
                                       out0, out0_h, hid_h, hid_c, TLEN);
  // ---- main biLSTM layer 1 ------------------------------------------------
  gemm(out0_h, Wih_h[2], bih[2], bhh[2], XPf, TLEN, G4, 1024);
  gemm(out0_h, Wih_h[3], bih[3], bhh[3], XPb, TLEN, G4, 1024);
  k_lstm_recur<<<2, 1024, 0, stream>>>(XPf, XPb, Whh_h[2], Whh_h[3],
                                       out_final, (_Float16*)nullptr,
                                       hid_h + 1024, hid_c + 1024, TLEN);

  // ---- copy_rep -----------------------------------------------------------
  k_copy_rep<<<4096, 256, 0, stream>>>(out_final, cidx, ccnt, crep);

  // ---- sentence path ------------------------------------------------------
  k_sent_gather<<<SLEN, 1024, 0, stream>>>(out_final, seps, sg_h);
  gemm(sg_h, w2sW_h, w2sb, (const float*)nullptr, sin_f, SLEN, 512, 1024);
  cvt(sin_f, sin_h, SLEN * 512);
  gemm(sin_h, sWih_h[0], sbih[0], sbhh[0], XPf, SLEN, G4, 512);
  gemm(sin_h, sWih_h[1], sbih[1], sbhh[1], XPb, SLEN, G4, 512);
  k_lstm_recur<<<2, 1024, 0, stream>>>(XPf, XPb, sWhh_h[0], sWhh_h[1],
                                       sout0, sout0_h, dump, dump + 1024, SLEN);
  gemm(sout0_h, sWih_h[2], sbih[2], sbhh[2], XPf, SLEN, G4, 1024);
  gemm(sout0_h, sWih_h[3], sbih[3], sbhh[3], XPb, SLEN, G4, 1024);
  k_lstm_recur<<<2, 1024, 0, stream>>>(XPf, XPb, sWhh_h[2], sWhh_h[3],
                                       sent_out, (_Float16*)nullptr,
                                       dump, dump + 1024, SLEN);
}
